// CasualSelfAttention_45981919871731
// MI455X (gfx1250) — matrix operand order repaired and checked by reading, not compiled, not run on hardware
//
#include <hip/hip_runtime.h>
#include <hip/hip_bf16.h>

// ---------------------------------------------------------------------------
// Causal self-attention for B=4, T=2048, C=1024, H=16 (hd=64) on gfx1250.
// Pipeline: fp32->bf16 pre-convert (x, W_attn, W_proj) -> QKV GEMM
// (bf16 WMMA, 64x64 tile/wave) -> flash attention (bf16 WMMA, online
// softmax) -> output projection GEMM (bf16 WMMA, fp32 out).
// All matmuls use v_wmma_f32_16x16x32_bf16 with fp32 accumulation.
// ---------------------------------------------------------------------------

typedef __attribute__((ext_vector_type(16))) __bf16 v16bf;
typedef __attribute__((ext_vector_type(8)))  __bf16 v8bf;
typedef __attribute__((ext_vector_type(8)))  float  v8f;

static constexpr int Bc = 4;
static constexpr int Tc = 2048;
static constexpr int Cc = 1024;
static constexpr int Hc = 16;
static constexpr int HD = 64;
static constexpr int Mrows = Bc * Tc;      // 8192 tokens
static constexpr int Nqkv  = 3 * Cc;       // 3072

__device__ __forceinline__ v8f wmma_bf16(v16bf a, v16bf b, v8f c) {
  // D = A(16x32 bf16) * B(32x16 bf16) + C(16x16 f32)
  return __builtin_amdgcn_wmma_f32_16x16x32_bf16(
      /*neg_a=*/false, a, /*neg_b=*/false, b,
      /*c_mod=*/(short)0, c, /*reuse_a=*/false, /*reuse_b=*/false);
}

__device__ __forceinline__ v8f zero8() {
  v8f r;
#pragma unroll
  for (int i = 0; i < 8; ++i) r[i] = 0.0f;
  return r;
}

__device__ __forceinline__ v16bf pack16(v8bf lo, v8bf hi) {
  v16bf r;
#pragma unroll
  for (int i = 0; i < 8; ++i) { r[i] = lo[i]; r[i + 8] = hi[i]; }
  return r;
}

// A-fragment (16x32) from a bf16 row; koff = (lane>>4)*8.
// half h holds K = 2*koff + h  -> two 16B chunks.
__device__ __forceinline__ v16bf load_a_bf16(const __bf16* __restrict__ row,
                                             int koff) {
  v8bf lo = *(const v8bf*)(row + 2 * koff);
  v8bf hi = *(const v8bf*)(row + 2 * koff + 8);
  return pack16(lo, hi);
}

// B-fragment (32x16) from bf16: p = W[n]*K + k + (lane>>4)*16; one 32B load.
__device__ __forceinline__ v16bf load_b_bf16(const __bf16* __restrict__ p) {
  return *(const v16bf*)p;
}

// ---------------------------------------------------------------------------
// Kernel 0: elementwise fp32 -> bf16 (8 elements per thread, b128 loads).
// ---------------------------------------------------------------------------
__global__ void __launch_bounds__(256)
cvt_f32_bf16(const float* __restrict__ src, __bf16* __restrict__ dst, int n) {
  int i = (blockIdx.x * 256 + threadIdx.x) * 8;
  if (i + 8 <= n) {
    v8bf o;
#pragma unroll
    for (int j = 0; j < 8; ++j) o[j] = (__bf16)src[i + j];
    *(v8bf*)(dst + i) = o;
  }
}

// ---------------------------------------------------------------------------
// Kernel 1: QKV = x @ W_attn^T (both bf16). 64x64 tile per wave:
// 16 accumulators, 256 B/lane of bf16 operands per 16 WMMAs.
// Q,K written [B,H,T,64] bf16; V written transposed [B,H,64,T] bf16.
// grid(48, 32), block 128 (4 waves).
// ---------------------------------------------------------------------------
__global__ void __launch_bounds__(128)
qkv_gemm(const __bf16* __restrict__ X, const __bf16* __restrict__ W,
         __bf16* __restrict__ Qw, __bf16* __restrict__ Kw,
         __bf16* __restrict__ Vt) {
  const int lane = threadIdx.x & 31;
  const int wave = threadIdx.x >> 5;
  const int mbase = blockIdx.y * 256 + wave * 64;
  const int nbase = blockIdx.x * 64;
  const int lm = lane & 15;
  const int hi = lane >> 4;
  const int koff = hi * 8;

  v8f acc[4][4];
#pragma unroll
  for (int i = 0; i < 4; ++i)
#pragma unroll
    for (int j = 0; j < 4; ++j) acc[i][j] = zero8();

  const __bf16* arow = X + (size_t)(mbase + lm) * Cc;
  const __bf16* brow = W + (size_t)(nbase + lm) * Cc + hi * 16;

  auto kstep = [&](int k) {
    v16bf af[4];
#pragma unroll
    for (int ai = 0; ai < 4; ++ai)
      af[ai] = load_a_bf16(arow + (size_t)ai * 16 * Cc + k, koff);
#pragma unroll
    for (int bi = 0; bi < 4; ++bi) {
      v16bf bf = load_b_bf16(brow + (size_t)bi * 16 * Cc + k);
#pragma unroll
      for (int ai = 0; ai < 4; ++ai)
        acc[ai][bi] = wmma_bf16(af[ai], bf, acc[ai][bi]);
    }
  };

#pragma unroll 1
  for (int k = 0; k < Cc - 32; k += 32) {
    // WGP-scope prefetch of the next k-step (global_prefetch_b8).
    __builtin_prefetch(arow + k + 32, 0, 3);
    __builtin_prefetch(brow + k + 32, 0, 3);
    kstep(k);
  }
  kstep(Cc - 32);  // peeled last step: no prefetch, no in-loop branch

  // Tile-uniform decode: a 64-wide aligned n-tile stays inside one of
  // {q,k,v} and inside one head; a 64-high aligned m-tile stays in one batch.
  const int which = nbase >> 10;                // 0=q 1=k 2=v
  const int hh = (nbase & (Cc - 1)) >> 6;       // head
  const int bidx = mbase >> 11;                 // batch
  const int t0 = mbase & (Tc - 1);              // token base within batch

  if (which < 2) {
    __bf16* base = (which == 0 ? Qw : Kw) +
                   (((size_t)bidx * Hc + hh) * Tc + t0) * HD;
#pragma unroll
    for (int ai = 0; ai < 4; ++ai)
#pragma unroll
      for (int bi = 0; bi < 4; ++bi)
#pragma unroll
        for (int j = 0; j < 8; ++j) {
          int tl = ai * 16 + j + 8 * hi;
          int d = bi * 16 + lm;
          base[(size_t)tl * HD + d] = (__bf16)acc[ai][bi][j];
        }
  } else {
    __bf16* base = Vt + ((size_t)bidx * Hc + hh) * HD * Tc + t0;
#pragma unroll
    for (int ai = 0; ai < 4; ++ai)
#pragma unroll
      for (int bi = 0; bi < 4; ++bi)
#pragma unroll
        for (int j = 0; j < 8; ++j) {
          int tl = ai * 16 + j + 8 * hi;
          int d = bi * 16 + lm;
          base[(size_t)d * Tc + tl] = (__bf16)acc[ai][bi][j];
        }
  }
}

// ---------------------------------------------------------------------------
// Kernel 2: flash attention. One wave per 16-row q-tile; 64-key blocks.
// grid(T/64, B*H), block 128 (4 waves). LDS: per-wave 16x64 bf16 P tile.
// ---------------------------------------------------------------------------
__global__ void __launch_bounds__(128)
flash_attn(const __bf16* __restrict__ Qw, const __bf16* __restrict__ Kw,
           const __bf16* __restrict__ Vt, __bf16* __restrict__ Yh) {
  __shared__ __bf16 plds[4][16 * 64];

  const int lane = threadIdx.x & 31;
  const int wave = threadIdx.x >> 5;
  const int lm = lane & 15;
  const int hi = lane >> 4;
  const int koff = hi * 8;

  const int q0 = blockIdx.x * 64 + wave * 16;
  const int bh = blockIdx.y;
  const int b = bh >> 4;
  const int h = bh & (Hc - 1);

  const __bf16* Qb = Qw + (size_t)bh * Tc * HD;
  const __bf16* Kb = Kw + (size_t)bh * Tc * HD;
  const __bf16* Vb = Vt + (size_t)bh * HD * Tc;
  __bf16* pl = plds[wave];

  // Q A-fragments: rows q0..q0+15, split over head dim (2 x K=32).
  v16bf qa[2];
#pragma unroll
  for (int s = 0; s < 2; ++s)
    qa[s] = load_a_bf16(Qb + (size_t)(q0 + lm) * HD + s * 32, koff);

  float mrow[8], lrow[8];
  v8f o[4];
#pragma unroll
  for (int j = 0; j < 8; ++j) { mrow[j] = -__builtin_inff(); lrow[j] = 0.0f; }
#pragma unroll
  for (int dt = 0; dt < 4; ++dt) o[dt] = zero8();

  const int qmax = q0 + 15;
  for (int kb = 0; kb <= qmax; kb += 64) {
    // WGP-scope prefetch of the next key block (lane-spread over rows).
    if (kb + 64 <= qmax) {
      __builtin_prefetch(Kb + (size_t)(kb + 64 + lane) * HD, 0, 3);
      __builtin_prefetch(Vb + (size_t)lane * Tc + kb + 64, 0, 3);
    }
    // ---- scores S = (Q K^T) * 0.125 over 64 keys -------------------------
    v8f s[4];
#pragma unroll
    for (int nt = 0; nt < 4; ++nt) s[nt] = zero8();
#pragma unroll
    for (int nt = 0; nt < 4; ++nt) {
      int key = kb + nt * 16 + lm;
#pragma unroll
      for (int ks = 0; ks < 2; ++ks) {
        v16bf kf = *(const v16bf*)(Kb + (size_t)key * HD + ks * 32 + hi * 16);
        s[nt] = wmma_bf16(qa[ks], kf, s[nt]);
      }
    }
    // ---- scale + causal mask --------------------------------------------
#pragma unroll
    for (int nt = 0; nt < 4; ++nt) {
      int key = kb + nt * 16 + lm;
#pragma unroll
      for (int j = 0; j < 8; ++j) {
        int q = q0 + j + 8 * hi;
        float v = s[nt][j] * 0.125f;
        s[nt][j] = (key > q) ? -__builtin_inff() : v;
      }
    }
    // ---- online softmax stats (row reduce across 16 lanes) --------------
    float bm[8], bs[8];
#pragma unroll
    for (int j = 0; j < 8; ++j) {
      float mx = s[0][j];
#pragma unroll
      for (int nt = 1; nt < 4; ++nt) mx = fmaxf(mx, s[nt][j]);
      bm[j] = mx;
    }
#pragma unroll
    for (int mask = 1; mask < 16; mask <<= 1)
#pragma unroll
      for (int j = 0; j < 8; ++j)
        bm[j] = fmaxf(bm[j], __shfl_xor(bm[j], mask, 32));

    float alpha[8];
#pragma unroll
    for (int j = 0; j < 8; ++j) {
      float nm = fmaxf(mrow[j], bm[j]);
      alpha[j] = __expf(mrow[j] - nm);
      mrow[j] = nm;
      bs[j] = 0.0f;
    }
#pragma unroll
    for (int nt = 0; nt < 4; ++nt)
#pragma unroll
      for (int j = 0; j < 8; ++j) {
        float p = __expf(s[nt][j] - mrow[j]);
        s[nt][j] = p;
        bs[j] += p;
      }
#pragma unroll
    for (int mask = 1; mask < 16; mask <<= 1)
#pragma unroll
      for (int j = 0; j < 8; ++j)
        bs[j] += __shfl_xor(bs[j], mask, 32);
#pragma unroll
    for (int j = 0; j < 8; ++j) lrow[j] = lrow[j] * alpha[j] + bs[j];
#pragma unroll
    for (int dt = 0; dt < 4; ++dt)
#pragma unroll
      for (int j = 0; j < 8; ++j) o[dt][j] *= alpha[j];

    // ---- P: C-layout -> A-layout through LDS (in-order per wave) --------
#pragma unroll
    for (int nt = 0; nt < 4; ++nt)
#pragma unroll
      for (int j = 0; j < 8; ++j)
        pl[(j + 8 * hi) * 64 + nt * 16 + lm] = (__bf16)s[nt][j];

    v16bf pa[2];
#pragma unroll
    for (int ks = 0; ks < 2; ++ks)
      pa[ks] = load_a_bf16(pl + lm * 64 + ks * 32, koff);

    // ---- O += P @ V ------------------------------------------------------
#pragma unroll
    for (int dt = 0; dt < 4; ++dt) {
      int d = dt * 16 + lm;
#pragma unroll
      for (int ks = 0; ks < 2; ++ks) {
        v16bf vf = *(const v16bf*)(Vb + (size_t)d * Tc + kb + ks * 32 + hi * 16);
        o[dt] = wmma_bf16(pa[ks], vf, o[dt]);
      }
    }
  }

  // ---- normalize and write head output [B,T,H*64] bf16 -------------------
#pragma unroll
  for (int j = 0; j < 8; ++j) lrow[j] = 1.0f / lrow[j];
#pragma unroll
  for (int dt = 0; dt < 4; ++dt)
#pragma unroll
    for (int j = 0; j < 8; ++j) {
      int t = q0 + j + 8 * hi;
      int d = dt * 16 + lm;
      Yh[((size_t)b * Tc + t) * Cc + h * HD + d] = (__bf16)(o[dt][j] * lrow[j]);
    }
}

// ---------------------------------------------------------------------------
// Kernel 3: out = Yh @ W_proj^T (bf16 in, fp32 out). 64x64 tile per wave.
// grid(16, 32), block 128.
// ---------------------------------------------------------------------------
__global__ void __launch_bounds__(128)
proj_gemm(const __bf16* __restrict__ Yh, const __bf16* __restrict__ W,
          float* __restrict__ out) {
  const int lane = threadIdx.x & 31;
  const int wave = threadIdx.x >> 5;
  const int mbase = blockIdx.y * 256 + wave * 64;
  const int nbase = blockIdx.x * 64;
  const int lm = lane & 15;
  const int hi = lane >> 4;
  const int koff = hi * 8;

  v8f acc[4][4];
#pragma unroll
  for (int i = 0; i < 4; ++i)
#pragma unroll
    for (int j = 0; j < 4; ++j) acc[i][j] = zero8();

  const __bf16* arow = Yh + (size_t)(mbase + lm) * Cc;
  const __bf16* brow = W + (size_t)(nbase + lm) * Cc + hi * 16;

  auto kstep = [&](int k) {
    v16bf af[4];
#pragma unroll
    for (int ai = 0; ai < 4; ++ai)
      af[ai] = load_a_bf16(arow + (size_t)ai * 16 * Cc + k, koff);
#pragma unroll
    for (int bi = 0; bi < 4; ++bi) {
      v16bf bf = load_b_bf16(brow + (size_t)bi * 16 * Cc + k);
#pragma unroll
      for (int ai = 0; ai < 4; ++ai)
        acc[ai][bi] = wmma_bf16(af[ai], bf, acc[ai][bi]);
    }
  };

#pragma unroll 1
  for (int k = 0; k < Cc - 32; k += 32) {
    __builtin_prefetch(arow + k + 32, 0, 3);
    __builtin_prefetch(brow + k + 32, 0, 3);
    kstep(k);
  }
  kstep(Cc - 32);

#pragma unroll
  for (int ai = 0; ai < 4; ++ai)
#pragma unroll
    for (int bi = 0; bi < 4; ++bi)
#pragma unroll
      for (int j = 0; j < 8; ++j) {
        int m = mbase + ai * 16 + j + 8 * hi;
        int n = nbase + bi * 16 + lm;
        out[(size_t)m * Cc + n] = acc[ai][bi][j];
      }
}

// ---------------------------------------------------------------------------
// Launcher. d_ws: Q|K|V^T|Yh (4 x 16MB) + x_bf (16MB) + Wa_bf (6MB) +
// Wp_bf (2MB) = 88MB.
// ---------------------------------------------------------------------------
extern "C" void kernel_launch(void* const* d_in, const int* in_sizes, int n_in,
                              void* d_out, int out_size, void* d_ws,
                              size_t ws_size, hipStream_t stream) {
  const float* x      = (const float*)d_in[0];
  const float* W_attn = (const float*)d_in[1];
  const float* W_proj = (const float*)d_in[2];
  float* out = (float*)d_out;

  const size_t elems = (size_t)Bc * Hc * Tc * HD;  // 8,388,608
  const int nx = Mrows * Cc;                        // 8,388,608
  const int nwa = Nqkv * Cc;                        // 3,145,728
  const int nwp = Cc * Cc;                          // 1,048,576

  __bf16* Qw  = (__bf16*)d_ws;
  __bf16* Kw  = Qw + elems;
  __bf16* Vt  = Kw + elems;
  __bf16* Yh  = Vt + elems;
  __bf16* Xb  = Yh + elems;
  __bf16* Wab = Xb + nx;
  __bf16* Wpb = Wab + nwa;

  // fp32 -> bf16 pre-conversion (HBM-cheap, removes cvt from GEMM hot loops).
  cvt_f32_bf16<<<(nx / 8 + 255) / 256, 256, 0, stream>>>(x, Xb, nx);
  cvt_f32_bf16<<<(nwa / 8 + 255) / 256, 256, 0, stream>>>(W_attn, Wab, nwa);
  cvt_f32_bf16<<<(nwp / 8 + 255) / 256, 256, 0, stream>>>(W_proj, Wpb, nwp);

  // QKV projection: M=8192, N=3072, K=1024 -> 64x64 tiles per wave.
  qkv_gemm<<<dim3(Nqkv / 64, Mrows / 256), 128, 0, stream>>>(Xb, Wab, Qw, Kw,
                                                             Vt);
  // Flash attention: one wave per 16-row q-tile.
  flash_attn<<<dim3(Tc / 64, Bc * Hc), 128, 0, stream>>>(Qw, Kw, Vt, Yh);
  // Output projection: M=8192, N=1024, K=1024 -> 64x64 tiles per wave.
  proj_gemm<<<dim3(Cc / 64, Mrows / 256), 128, 0, stream>>>(Yh, Wpb, out);
}
